// LEDMambaHypergraphSummariser_32246614459131
// MI455X (gfx1250) — matrix operand order repaired and checked by reading, not compiled, not run on hardware
//
#include <hip/hip_runtime.h>
#include <hip/hip_bf16.h>

typedef __attribute__((ext_vector_type(8)))  __bf16 v8bf;
typedef __attribute__((ext_vector_type(16))) __bf16 v16bf;
typedef __attribute__((ext_vector_type(8)))  float  v8f;
typedef __attribute__((ext_vector_type(4)))  unsigned int v4u;
typedef __attribute__((ext_vector_type(8)))  int v8i_t;
typedef __attribute__((ext_vector_type(4)))  int v4i_t;

#define BN_       200
#define S_        128
#define D_        1024
#define DIN_      2048
#define DSTATE_   32
#define ROWS_     (BN_ * S_)          // 25600
#define NPROJ_    (2 * DSTATE_ + 1)   // 65

// ---------------------------------------------------------------------------
// LayerNorm (f32) + cast to bf16 : one block per (bn,s) row of D=1024
// ---------------------------------------------------------------------------
__global__ __launch_bounds__(256) void ln_cast_kernel(
    const float* __restrict__ x, const float* __restrict__ w,
    const float* __restrict__ b, __bf16* __restrict__ out)
{
    __shared__ float red[256];
    const int row = blockIdx.x;
    const float* xr = x + (size_t)row * D_;
    float s = 0.f, ss = 0.f;
    for (int i = threadIdx.x; i < D_; i += 256) {
        float v = xr[i]; s += v; ss += v * v;
    }
    red[threadIdx.x] = s; __syncthreads();
    for (int o = 128; o > 0; o >>= 1) {
        if (threadIdx.x < o) red[threadIdx.x] += red[threadIdx.x + o];
        __syncthreads();
    }
    const float mean = red[0] * (1.f / D_);
    __syncthreads();
    red[threadIdx.x] = ss; __syncthreads();
    for (int o = 128; o > 0; o >>= 1) {
        if (threadIdx.x < o) red[threadIdx.x] += red[threadIdx.x + o];
        __syncthreads();
    }
    const float var  = red[0] * (1.f / D_) - mean * mean;   // biased variance
    const float rstd = rsqrtf(var + 1e-5f);
    for (int i = threadIdx.x; i < D_; i += 256) {
        float v = (xr[i] - mean) * rstd * w[i] + b[i];
        out[(size_t)row * D_ + i] = (__bf16)v;
    }
}

// ---------------------------------------------------------------------------
// f32 -> bf16 elementwise cast (weights)
// ---------------------------------------------------------------------------
__global__ __launch_bounds__(256) void cast_bf16_kernel(
    const float* __restrict__ in, __bf16* __restrict__ out, int n)
{
    int i = blockIdx.x * 256 + threadIdx.x;
    if (i < n) out[i] = (__bf16)in[i];
}

// ---------------------------------------------------------------------------
// Tensor Data Mover: DMA a 128-row x 32-col bf16 tile (row stride = K elems)
// from global memory into LDS at byte offset lds_off. 2D descriptor:
//   group0: count=1 | lds_addr | global_addr[56:0] | type=2
//   group1: data_size=1(2B), tensor_dim0=K, tensor_dim1=rows,
//           tile_dim0=32, tile_dim1=128, tensor_dim0_stride=K
// Tracked by TENSORcnt; issued by one wave, completion fanned out via barrier.
// ---------------------------------------------------------------------------
__device__ __forceinline__ void tdm_load_tile_128x32(
    unsigned lds_off, const __bf16* gptr, int K, int rows)
{
    const unsigned long long ga = (unsigned long long)(uintptr_t)gptr;
    v4u g0;
    g0[0] = 1u;                                        // count = 1 descriptor
    g0[1] = lds_off;                                   // LDS byte address
    g0[2] = (unsigned)(ga & 0xffffffffu);              // global_addr[31:0]
    g0[3] = (unsigned)((ga >> 32) & 0x1ffffffu)        // global_addr[56:32]
          | (2u << 30);                                // type = 2 ("image")
    v8i_t g1;
    g1[0] = (int)(1u << 16);                           // data_size=1 (2 bytes)
    g1[1] = (int)(((unsigned)K & 0xffffu) << 16);      // tensor_dim0[15:0]
    g1[2] = (int)((((unsigned)K >> 16) & 0xffffu)      // tensor_dim0[31:16]
          | (((unsigned)rows & 0xffffu) << 16));       // tensor_dim1[15:0]
    g1[3] = (int)((((unsigned)rows >> 16) & 0xffffu)   // tensor_dim1[31:16]
          | (32u << 16));                              // tile_dim0 = 32
    g1[4] = 128;                                       // tile_dim1 = 128
    g1[5] = K;                                         // tensor_dim0_stride
    g1[6] = 0;
    g1[7] = 0;
    const v4i_t z4 = {0, 0, 0, 0};
    const v8i_t z8 = {0, 0, 0, 0, 0, 0, 0, 0};
    __builtin_amdgcn_tensor_load_to_lds(g0, g1, z4, z4, z8, 0);
}

// ---------------------------------------------------------------------------
// Tiled bf16 WMMA GEMM:  C[m,n] = sum_k A[m,k]*Bw[n,k]  (+ residual)
// Block: 256 thr = 8 waves (4M x 2N), block tile 128M x 128N.
// Wave tile: 32M x 64N = 8 f32 accumulators (2 A-frags x 4 B-frags per K-step).
// TDM double-buffers 128x32 A/B tiles in LDS (32KB), overlapping DMA with
// v_wmma_f32_16x16x32_bf16 issue; sync = s_wait_tensorcnt + workgroup barrier.
// ---------------------------------------------------------------------------
__global__ __launch_bounds__(256) void gemm_bf16_wmma_kernel(
    const __bf16* __restrict__ A, const __bf16* __restrict__ Bw,
    const float* __restrict__ Res, float* __restrict__ C,
    int M, int N, int K)
{
    __shared__ __bf16 lds_a[2][128 * 32];   // [buf][m_local][k] 8KB each
    __shared__ __bf16 lds_b[2][128 * 32];   // [buf][n_local][k] 8KB each

    const int tid   = threadIdx.x;
    const int wave  = tid >> 5;          // 0..7
    const int lane  = tid & 31;
    const int wm    = wave & 3;          // M strip 0..3 (32 rows each)
    const int wn    = wave >> 2;         // N strip 0..1 (64 cols each)
    const int m_tile0 = blockIdx.y * 128;
    const int n_tile0 = blockIdx.x * 128;

    const int lane15 = lane & 15;
    const int hi     = lane >> 4;        // lanes 16-31 hold the K/M upper half
    const int ka     = hi * 8;           // A k-base per ISA 16-bit A layout
    const int kb     = hi * 8;           // B k-base per ISA 16-bit B layout

    v8f acc[8] = {};                     // acc[i*4+j]: i = M sub-tile, j = N sub-tile

    // prologue: DMA buffer 0 (wave 0 issues, everyone waits at the barrier)
    if (wave == 0) {
        tdm_load_tile_128x32((unsigned)(uintptr_t)&lds_a[0][0],
                             A  + (size_t)m_tile0 * K, K, 128);
        tdm_load_tile_128x32((unsigned)(uintptr_t)&lds_b[0][0],
                             Bw + (size_t)n_tile0 * K, K, 128);
        __builtin_amdgcn_s_wait_tensorcnt(0);
    }
    __syncthreads();

    int cur = 0;
    for (int k0 = 0; k0 < K; k0 += 32) {
        // kick off DMA of the next K-slice into the other buffer
        if (wave == 0 && k0 + 32 < K) {
            tdm_load_tile_128x32((unsigned)(uintptr_t)&lds_a[cur ^ 1][0],
                                 A  + (size_t)m_tile0 * K + k0 + 32, K, 128);
            tdm_load_tile_128x32((unsigned)(uintptr_t)&lds_b[cur ^ 1][0],
                                 Bw + (size_t)n_tile0 * K + k0 + 32, K, 128);
        }

        // A fragments for this wave's two 16-row sub-tiles
        v16bf a0, a1;
        {
            const int ar0 = (wm * 32 + lane15) * 32;
            const int ar1 = (wm * 32 + 16 + lane15) * 32;
            v8bf x0 = *(const v8bf*)&lds_a[cur][ar0 + ka];
            v8bf x1 = *(const v8bf*)&lds_a[cur][ar0 + ka + 16];
            v8bf y0 = *(const v8bf*)&lds_a[cur][ar1 + ka];
            v8bf y1 = *(const v8bf*)&lds_a[cur][ar1 + ka + 16];
            #pragma unroll
            for (int i = 0; i < 8; i++) {
                a0[i] = x0[i]; a0[i + 8] = x1[i];
                a1[i] = y0[i]; a1[i + 8] = y1[i];
            }
        }
        // four 16-col B sub-tiles, two WMMAs each
        #pragma unroll
        for (int j = 0; j < 4; j++) {
            const int br = (wn * 64 + j * 16 + lane15) * 32;
            v16bf bfrag;
            v8bf b0 = *(const v8bf*)&lds_b[cur][br + kb];
            v8bf b1 = *(const v8bf*)&lds_b[cur][br + kb + 16];
            #pragma unroll
            for (int i = 0; i < 8; i++) { bfrag[i] = b0[i]; bfrag[i + 8] = b1[i]; }
            acc[j]     = __builtin_amdgcn_wmma_f32_16x16x32_bf16(
                false, a0, false, bfrag, (short)0, acc[j],     false, false);
            acc[4 + j] = __builtin_amdgcn_wmma_f32_16x16x32_bf16(
                false, a1, false, bfrag, (short)0, acc[4 + j], false, false);
        }

        // next buffer must be resident before anyone proceeds
        if (wave == 0 && k0 + 32 < K)
            __builtin_amdgcn_s_wait_tensorcnt(0);
        __syncthreads();
        cur ^= 1;
    }

    // store: row = r + 8*hi, col = lane&15  (16x16 f32 C/D layout)
    #pragma unroll
    for (int i = 0; i < 2; i++) {
        #pragma unroll
        for (int j = 0; j < 4; j++) {
            const v8f a = acc[i * 4 + j];
            const int n = n_tile0 + wn * 64 + j * 16 + lane15;
            const int mrow = m_tile0 + wm * 32 + i * 16 + 8 * hi;
            #pragma unroll
            for (int r = 0; r < 8; r++) {
                const size_t idx = (size_t)(mrow + r) * N + n;
                float v = a[r];
                if (Res) v += Res[idx];
                C[idx] = v;
            }
        }
    }
}

// ---------------------------------------------------------------------------
// Causal depthwise conv (4 taps over S) + SiLU. thread = (bn, channel)
// ---------------------------------------------------------------------------
__global__ __launch_bounds__(256) void conv_silu_kernel(
    const float* __restrict__ xz, const float* __restrict__ cw,
    const float* __restrict__ cb, float* __restrict__ xact)
{
    const int t  = blockIdx.x * 256 + threadIdx.x;    // 0..409599
    const int c  = t & (DIN_ - 1);
    const int bn = t >> 11;
    const float w0 = cw[c * 4 + 0], w1 = cw[c * 4 + 1];
    const float w2 = cw[c * 4 + 2], w3 = cw[c * 4 + 3];
    const float bias = cb[c];
    float xm3 = 0.f, xm2 = 0.f, xm1 = 0.f;
    const size_t base = (size_t)bn * S_ * (2 * DIN_) + c;   // x_in = xz[:, 0:DIN]
    for (int s = 0; s < S_; s++) {
        const float x0 = xz[base + (size_t)s * (2 * DIN_)];
        const float acc = w0 * xm3 + w1 * xm2 + w2 * xm1 + w3 * x0 + bias;
        const float sv  = acc / (1.f + __expf(-acc));       // SiLU
        xact[((size_t)bn * S_ + s) * DIN_ + c] = sv;
        xm3 = xm2; xm2 = xm1; xm1 = x0;
    }
}

// ---------------------------------------------------------------------------
// x_proj: x_ssm[m,0:65] = x_act[m,:] @ x_proj_W.T ; split into draw/B/C
// one block per row m; row cached in LDS
// ---------------------------------------------------------------------------
__global__ __launch_bounds__(256) void xproj_kernel(
    const float* __restrict__ xact, const float* __restrict__ W,
    float* __restrict__ draw, float* __restrict__ Bm, float* __restrict__ Cm)
{
    __shared__ float row[DIN_];
    const int m = blockIdx.x;
    const float* xr = xact + (size_t)m * DIN_;
    for (int i = threadIdx.x; i < DIN_; i += 256) row[i] = xr[i];
    __syncthreads();
    const int o = threadIdx.x;
    if (o < NPROJ_) {
        const float* wr = W + (size_t)o * DIN_;
        float acc = 0.f;
        for (int k = 0; k < DIN_; k++) acc = fmaf(row[k], wr[k], acc);
        if (o == 0)       draw[m] = acc;
        else if (o < 33)  Bm[(size_t)m * DSTATE_ + (o - 1)]  = acc;
        else              Cm[(size_t)m * DSTATE_ + (o - 33)] = acc;
    }
}

// ---------------------------------------------------------------------------
// Sequential selective scan. thread = (bn, d); h[32] in VGPRs, B_t/C_t via LDS.
// Delta computed on the fly (saves a 210MB tensor). Emits gated bf16 output.
// ---------------------------------------------------------------------------
__global__ __launch_bounds__(256) void scan_kernel(
    const float* __restrict__ draw, const float* __restrict__ Bm,
    const float* __restrict__ Cm,   const float* __restrict__ xact,
    const float* __restrict__ xz,   const float* __restrict__ dtW,
    const float* __restrict__ dtb,  const float* __restrict__ eb,
    const float* __restrict__ escale, __bf16* __restrict__ yg)
{
    __shared__ float Bs[DSTATE_], Cs[DSTATE_];
    const int bn = blockIdx.x >> 3;
    const int d  = ((blockIdx.x & 7) << 8) + threadIdx.x;
    const float w  = dtW[d];
    const float bb = dtb[d];
    const float es = escale[0];
    float h[DSTATE_];
    #pragma unroll
    for (int n = 0; n < DSTATE_; n++) h[n] = 0.f;

    for (int s = 0; s < S_; s++) {
        const size_t m = (size_t)bn * S_ + s;
        if (threadIdx.x < 32)      Bs[threadIdx.x]      = Bm[m * DSTATE_ + threadIdx.x];
        else if (threadIdx.x < 64) Cs[threadIdx.x - 32] = Cm[m * DSTATE_ + threadIdx.x - 32];
        __syncthreads();

        const float t  = fmaf(draw[m], w, bb);
        const float sp = (t > 20.f) ? t : log1pf(__expf(t));     // softplus
        const float dt = fmaxf(fmaf(es, eb[m], sp), 1e-4f);
        const float a  = __expf(-dt);
        const float xa = xact[m * DIN_ + d];

        float y = 0.f;
        #pragma unroll
        for (int n = 0; n < DSTATE_; n++) {
            float hv = fmaf(h[n], a, Bs[n] * xa);
            hv = fminf(fmaxf(hv, -100.f), 100.f);
            h[n] = hv;
            y = fmaf(hv, Cs[n], y);
        }
        const float z = xz[m * (2 * DIN_) + DIN_ + d];           // gate input
        const float g = z / (1.f + __expf(-z));                  // SiLU(z)
        yg[m * DIN_ + d] = (__bf16)(y * g);
        __syncthreads();
    }
}

// ---------------------------------------------------------------------------
extern "C" void kernel_launch(void* const* d_in, const int* in_sizes, int n_in,
                              void* d_out, int out_size, void* d_ws, size_t ws_size,
                              hipStream_t stream)
{
    const float* x        = (const float*)d_in[0];
    const float* ebias    = (const float*)d_in[1];
    const float* norm_w   = (const float*)d_in[2];
    const float* norm_b   = (const float*)d_in[3];
    const float* in_projW = (const float*)d_in[4];
    const float* conv_w   = (const float*)d_in[5];
    const float* conv_b   = (const float*)d_in[6];
    const float* x_projW  = (const float*)d_in[7];
    const float* dt_W     = (const float*)d_in[8];
    const float* dt_b     = (const float*)d_in[9];
    const float* out_W    = (const float*)d_in[10];
    const float* escale   = (const float*)d_in[11];
    float* out = (float*)d_out;

    // workspace carve-up (256B aligned)
    char* ws = (char*)d_ws;
    size_t off = 0;
    auto carve = [&](size_t bytes) {
        void* p = ws + off;
        off += (bytes + 255) & ~(size_t)255;
        return p;
    };
    __bf16* xn_bf   = (__bf16*)carve((size_t)ROWS_ * D_ * 2);        // 52.4 MB
    __bf16* win_bf  = (__bf16*)carve((size_t)2 * DIN_ * D_ * 2);     //  8.4 MB
    __bf16* wout_bf = (__bf16*)carve((size_t)D_ * DIN_ * 2);         //  4.2 MB
    float*  xz      = (float*)carve((size_t)ROWS_ * 2 * DIN_ * 4);   //  419 MB
    float*  xact    = (float*)carve((size_t)ROWS_ * DIN_ * 4);       //  210 MB
    float*  draw    = (float*)carve((size_t)ROWS_ * 4);
    float*  Bmat    = (float*)carve((size_t)ROWS_ * DSTATE_ * 4);
    float*  Cmat    = (float*)carve((size_t)ROWS_ * DSTATE_ * 4);
    __bf16* yg_bf   = (__bf16*)carve((size_t)ROWS_ * DIN_ * 2);      //  105 MB

    // 1. LayerNorm + bf16 cast
    ln_cast_kernel<<<ROWS_, 256, 0, stream>>>(x, norm_w, norm_b, xn_bf);

    // 2. weight casts
    {
        const int n1 = 2 * DIN_ * D_;      // 4,194,304
        cast_bf16_kernel<<<(n1 + 255) / 256, 256, 0, stream>>>(in_projW, win_bf, n1);
        const int n2 = D_ * DIN_;          // 2,097,152
        cast_bf16_kernel<<<(n2 + 255) / 256, 256, 0, stream>>>(out_W, wout_bf, n2);
    }

    // 3. GEMM1: xz = xn @ in_proj_W.T   (M=25600, N=4096, K=1024)
    gemm_bf16_wmma_kernel<<<dim3((2 * DIN_) / 128, ROWS_ / 128), 256, 0, stream>>>(
        xn_bf, win_bf, nullptr, xz, ROWS_, 2 * DIN_, D_);

    // 4. depthwise conv + SiLU
    conv_silu_kernel<<<(BN_ * DIN_) / 256, 256, 0, stream>>>(xz, conv_w, conv_b, xact);

    // 5. x_proj split
    xproj_kernel<<<ROWS_, 256, 0, stream>>>(xact, x_projW, draw, Bmat, Cmat);

    // 6. selective scan + gate (bf16 out)
    scan_kernel<<<BN_ * (DIN_ / 256), 256, 0, stream>>>(
        draw, Bmat, Cmat, xact, xz, dt_W, dt_b, ebias, escale, yg_bf);

    // 7. GEMM2 + residual: out = yg @ out_W.T + x   (M=25600, N=1024, K=2048)
    gemm_bf16_wmma_kernel<<<dim3(D_ / 128, ROWS_ / 128), 256, 0, stream>>>(
        yg_bf, wout_bf, x, out, ROWS_, D_, DIN_);
}
